// WCSA_36318243455396
// MI455X (gfx1250) — matrix operand order repaired and checked by reading, not compiled, not run on hardware
//
#include <hip/hip_runtime.h>

// ---------------------------------------------------------------------------
// Types for CDNA5 WMMA (wave32): bf16 A/B fragments (16 x bf16 = 8 VGPRs),
// f32 accumulator (8 x f32 = 8 VGPRs).
// ---------------------------------------------------------------------------
typedef __bf16 bf16_t;
typedef __attribute__((ext_vector_type(8)))  __bf16 bf16x8;
typedef __attribute__((ext_vector_type(16))) __bf16 bf16x16;
typedef __attribute__((ext_vector_type(8)))  float  f32x8;
typedef __attribute__((ext_vector_type(4)))  unsigned int u32x4;
typedef __attribute__((ext_vector_type(8)))  int i32x8;
typedef __attribute__((ext_vector_type(4)))  int i32x4;

union FragAB {
    bf16x16        v;
    bf16x8         h[2];
    unsigned short u[16];
};

__device__ __forceinline__ unsigned short f32_bf16(float f) {
    unsigned u = __float_as_uint(f);
    unsigned r = u + 0x7FFFu + ((u >> 16) & 1u);   // round-to-nearest-even
    return (unsigned short)(r >> 16);
}

__device__ __forceinline__ f32x8 wmma_bf16(bf16x16 a, bf16x16 b, f32x8 c) {
    // 8 args: (neg_a, A, neg_b, B, c_mod, C, reuse_a, reuse_b)
    return __builtin_amdgcn_wmma_f32_16x16x32_bf16(false, a, false, b,
                                                   (short)0, c, false, false);
}

// A fragment, 16x32 (MxK), source row-major with leading dim `ld` (elements).
// Lane (m = lane&15, h = lane>>4) holds row m, K = [h*8, h*8+7] and [16+h*8, ...].
template <typename PT>
__device__ __forceinline__ bf16x16 load_a_rm(PT p0, int ld) {
    int lane = threadIdx.x & 31;
    int m = lane & 15, hp = lane >> 4;
    FragAB f;
    f.h[0] = *(const bf16x8*)(p0 + m * ld + hp * 8);
    f.h[1] = *(const bf16x8*)(p0 + m * ld + hp * 8 + 16);
    return f.v;
}

// B fragment, 32x16 (KxN), where B[k][n] is stored N-major: source row n has
// contiguous K (e.g. weight W[n][k] row-major, or K/V^T tiles in LDS).
// Lane (n = lane&15, h = lane>>4) holds col n, K = [h*8, h*8+7] and [16+h*8, ...].
template <typename PT>
__device__ __forceinline__ bf16x16 load_b_nk(PT p0, int ld) {
    int lane = threadIdx.x & 31;
    int n = lane & 15, hp = lane >> 4;
    FragAB f;
    f.h[0] = *(const bf16x8*)(p0 + n * ld + hp * 8);
    f.h[1] = *(const bf16x8*)(p0 + n * ld + hp * 8 + 16);
    return f.v;
}

// ---------------------------------------------------------------------------
// Tensor Data Mover: 2D tile load (bf16 elements) global -> LDS.
// D# per CDNA5 ISA:
//   group0: [1:0]=count=1, [63:32]=lds_addr, [120:64]=global_addr, type=2
//   group1: data_size=1 (2B), tensor_dim0/1 = tile dims (no OOB),
//           tile_dim0 = elems/row, tile_dim1 = rows, tensor_dim0_stride = pitch
// Issued by a single wave; tracked with TENSORcnt.
// 6-arg builtin (this toolchain): (g0, g1, g2, g3, g_extra, cpol)
// ---------------------------------------------------------------------------
__device__ __forceinline__ void tdm_load_2d_bf16(void* lds_dst, const void* gsrc,
                                                 unsigned tile_w, unsigned tile_h,
                                                 unsigned row_stride_elems) {
    unsigned lds_off = (unsigned)(unsigned long long)lds_dst;  // LDS byte offset
    unsigned long long ga = (unsigned long long)gsrc;
    u32x4 g0;
    g0[0] = 1u;                                              // count=1, user desc
    g0[1] = lds_off;                                         // lds_addr (bytes)
    g0[2] = (unsigned)(ga & 0xFFFFFFFFu);                    // global_addr[31:0]
    g0[3] = (unsigned)((ga >> 32) & 0x1FFFFFFu) | (2u << 30); // addr[56:32] | type=2
    i32x8 g1;
    g1[0] = (int)(1u << 16);                                 // data_size=1 (2 bytes)
    g1[1] = (int)((tile_w & 0xFFFFu) << 16);                 // tensor_dim0 lo16
    g1[2] = (int)(((tile_w >> 16) & 0xFFFFu) | ((tile_h & 0xFFFFu) << 16));
    g1[3] = (int)(((tile_h >> 16) & 0xFFFFu) | (tile_w << 16)); // tile_dim0
    g1[4] = (int)(tile_h & 0xFFFFu);                         // tile_dim1 (tile_dim2=0)
    g1[5] = (int)row_stride_elems;                           // tensor_dim0_stride lo32
    g1[6] = 0;                                               // stride hi / dim1_stride lo
    g1[7] = 0;
    i32x4 z4 = {0, 0, 0, 0};
    i32x8 z8 = {0, 0, 0, 0, 0, 0, 0, 0};
    __builtin_amdgcn_tensor_load_to_lds(g0, g1, z4, z4, z8, 0);
}

// ---------------------------------------------------------------------------
// Kernel 1: f32 -> bf16 conversion (activations and weights)
// ---------------------------------------------------------------------------
__global__ void k_cvt_bf16(const float* __restrict__ in,
                           unsigned short* __restrict__ out, int n) {
    int i = blockIdx.x * 256 + threadIdx.x;
    if (i < n) out[i] = f32_bf16(in[i]);
}

// ---------------------------------------------------------------------------
// Kernel 2: projection GEMM  Cout[M x Nc] = A[M x K] @ W[Nc x K]^T  (bf16 out)
// 256 threads = 8 waves; block tile 128x64; each wave 32x32 (2x2 WMMA tiles).
// ---------------------------------------------------------------------------
__global__ __launch_bounds__(256) void k_proj_gemm(
    const unsigned short* __restrict__ A, const unsigned short* __restrict__ W,
    unsigned short* __restrict__ Cout, int M, int Nc, int K) {
    const bf16_t* Ab = (const bf16_t*)A;
    const bf16_t* Wb = (const bf16_t*)W;
    int w = threadIdx.x >> 5, lane = threadIdx.x & 31;
    int row0 = blockIdx.x * 128 + (w & 3) * 32;
    int col0 = blockIdx.y * 64 + (w >> 2) * 32;
    f32x8 zero = {0.f, 0.f, 0.f, 0.f, 0.f, 0.f, 0.f, 0.f};
    f32x8 acc[2][2] = {{zero, zero}, {zero, zero}};
    for (int k0 = 0; k0 < K; k0 += 32) {
        bf16x16 a0 = load_a_rm(Ab + (size_t)(row0 +  0) * K + k0, K);
        bf16x16 a1 = load_a_rm(Ab + (size_t)(row0 + 16) * K + k0, K);
        bf16x16 b0 = load_b_nk(Wb + (size_t)(col0 +  0) * K + k0, K);
        bf16x16 b1 = load_b_nk(Wb + (size_t)(col0 + 16) * K + k0, K);
        acc[0][0] = wmma_bf16(a0, b0, acc[0][0]);
        acc[0][1] = wmma_bf16(a0, b1, acc[0][1]);
        acc[1][0] = wmma_bf16(a1, b0, acc[1][0]);
        acc[1][1] = wmma_bf16(a1, b1, acc[1][1]);
    }
    int m = lane & 15, hp = lane >> 4;
#pragma unroll
    for (int i = 0; i < 2; i++)
#pragma unroll
        for (int j = 0; j < 2; j++) {
            int rbase = row0 + i * 16 + hp * 8;
            int c = col0 + j * 16 + m;
#pragma unroll
            for (int r = 0; r < 8; r++)
                Cout[(size_t)(rbase + r) * Nc + c] = f32_bf16(acc[i][j][r]);
        }
}

// ---------------------------------------------------------------------------
// Kernel 3: channel-branch scores. Per (b,h): S = q_c^T-slice @ k_c-slice over
// N=2048, softmax(rows, * scale*temp), store bf16 S (64x64) to workspace.
// ---------------------------------------------------------------------------
__global__ __launch_bounds__(256) void k_chan_scores(
    const unsigned short* __restrict__ Pq, const unsigned short* __restrict__ Pk,
    const float* __restrict__ temp, unsigned short* __restrict__ Sca,
    int N, int C) {
    __shared__ float Ssm[64 * 64];
    int bh = blockIdx.x, b = bh >> 2, hd = bh & 3;
    int w = threadIdx.x >> 5, lane = threadIdx.x & 31;
    int m = lane & 15, hp = lane >> 4;
    int t0 = w * 2;
    int ti = t0 >> 2, tj0 = t0 & 3;
    int cq  = hd * 64 + ti * 16 + m;
    int ckA = hd * 64 + tj0 * 16 + m;
    int ckB = ckA + 16;
    const unsigned short* Pqb = Pq + (size_t)b * N * C;
    const unsigned short* Pkb = Pk + (size_t)b * N * C;
    f32x8 acc0 = {0.f,0.f,0.f,0.f,0.f,0.f,0.f,0.f};
    f32x8 acc1 = acc0;
    for (int n0 = 0; n0 < N; n0 += 32) {
        FragAB fa, fb0, fb1;
#pragma unroll
        for (int e = 0; e < 16; e++) {
            int kk = (e < 8) ? (hp * 8 + e) : (16 + hp * 8 + (e - 8));
            fa.u[e] = Pqb[(size_t)(n0 + kk) * C + cq];
        }
#pragma unroll
        for (int e = 0; e < 16; e++) {
            int kk = (e < 8) ? (hp * 8 + e) : (16 + hp * 8 + (e - 8));
            fb0.u[e] = Pkb[(size_t)(n0 + kk) * C + ckA];
            fb1.u[e] = Pkb[(size_t)(n0 + kk) * C + ckB];
        }
        acc0 = wmma_bf16(fa.v, fb0.v, acc0);
        acc1 = wmma_bf16(fa.v, fb1.v, acc1);
    }
#pragma unroll
    for (int r = 0; r < 8; r++) {
        int i = ti * 16 + r + hp * 8;
        Ssm[i * 64 + tj0 * 16 + m]        = acc0[r];
        Ssm[i * 64 + (tj0 + 1) * 16 + m]  = acc1[r];
    }
    __syncthreads();
    if (threadIdx.x < 64) {
        int r = threadIdx.x;
        float st = 0.125f * temp[hd];
        float mx = -1e30f;
        for (int c = 0; c < 64; c++) {
            float v = Ssm[r * 64 + c] * st;
            Ssm[r * 64 + c] = v;
            mx = fmaxf(mx, v);
        }
        float sum = 0.f;
        for (int c = 0; c < 64; c++) {
            float p = __expf(Ssm[r * 64 + c] - mx);
            Ssm[r * 64 + c] = p;
            sum += p;
        }
        float inv = 1.f / sum;
        unsigned short* o = Sca + (size_t)bh * 64 * 64 + (size_t)r * 64;
        for (int c = 0; c < 64; c++) o[c] = f32_bf16(Ssm[r * 64 + c] * inv);
    }
}

// ---------------------------------------------------------------------------
// Kernel 4: channel-branch output  X = S(64x64) @ V^T(64xN), raw-reshape into
// d_out columns [0,256).
// ---------------------------------------------------------------------------
__global__ __launch_bounds__(256) void k_chan_av(
    const unsigned short* __restrict__ Sca, const unsigned short* __restrict__ Pv,
    float* __restrict__ out, int N, int C) {
    int bh = blockIdx.y, b = bh >> 2, hd = bh & 3;
    int n0 = blockIdx.x * 128;
    int w = threadIdx.x >> 5, lane = threadIdx.x & 31;
    int m = lane & 15, hp = lane >> 4;
    int ti = w & 3;
    int cj0 = (w >> 2) * 64;
    const bf16_t* S = (const bf16_t*)(Sca + (size_t)bh * 64 * 64);
    const bf16_t* Pvb = (const bf16_t*)Pv + (size_t)b * N * C;
    f32x8 zero = {0.f,0.f,0.f,0.f,0.f,0.f,0.f,0.f};
    f32x8 acc[4] = {zero, zero, zero, zero};
    for (int k0 = 0; k0 < 64; k0 += 32) {
        bf16x16 a = load_a_rm(S + ti * 16 * 64 + k0, 64);
#pragma unroll
        for (int j = 0; j < 4; j++) {
            bf16x16 bb = load_b_nk(
                Pvb + (size_t)(n0 + cj0 + j * 16) * C + hd * 64 + k0, C);
            acc[j] = wmma_bf16(a, bb, acc[j]);
        }
    }
#pragma unroll
    for (int j = 0; j < 4; j++) {
#pragma unroll
        for (int r = 0; r < 8; r++) {
            int i = ti * 16 + r + hp * 8;
            int n = n0 + cj0 + j * 16 + m;
            long flat = ((long)hd * 64 + i) * N + n;   // torch raw reshape
            int rp = (int)(flat >> 8);
            int c  = (int)(flat & 255);
            out[((size_t)b * N + rp) * 320 + c] = acc[j][r];
        }
    }
}

// ---------------------------------------------------------------------------
// Kernel 5: spatial flash attention with TDM staging.
// 64 query rows / workgroup. Wave 0 DMAs Q once and double-buffers the K
// (64x64) and V (64x16) blocks with tensor_load_to_lds; compute on block i
// overlaps the DMA of block i+1. QK^T and P@V via WMMA from LDS.
// ---------------------------------------------------------------------------
__global__ __launch_bounds__(256) void k_spatial_flash(
    const unsigned short* __restrict__ Pq, const unsigned short* __restrict__ Pk,
    const unsigned short* __restrict__ Pv, const float* __restrict__ temp2,
    float* __restrict__ out, int N, int C, int Cs) {
    __shared__ unsigned short Qs[64 * 64];        // 8 KB  Q tile
    __shared__ unsigned short Kblk[2][64 * 64];   // 16 KB double-buffered K
    __shared__ unsigned short Vraw[2][64 * 16];   // 4 KB  double-buffered V
    __shared__ unsigned short Vt[16 * 64];        // 2 KB  V^T (j, k)
    __shared__ float          Sblk[64 * 64];      // 16 KB scores
    __shared__ unsigned short Pblk[64 * 64];      // 8 KB  exp(P) bf16
    __shared__ float          Oacc[64 * 16];      // 4 KB  output accumulator
    __shared__ float          Mrow[64], Lrow[64];
    int bh = blockIdx.y, b = bh >> 2, hd = bh & 3;
    int q0 = blockIdx.x * 64;
    int w = threadIdx.x >> 5, lane = threadIdx.x & 31;
    int m = lane & 15, hp = lane >> 4;
    float st = 0.125f * temp2[hd];
    const bf16_t* Qb = (const bf16_t*)Pq + (size_t)b * N * C + hd * 64;
    const bf16_t* Kb = (const bf16_t*)Pk + (size_t)b * N * C + hd * 64;
    const bf16_t* Vb = (const bf16_t*)Pv + (size_t)b * N * Cs + hd * 16;

    if (threadIdx.x < 64) {
        Mrow[threadIdx.x] = -1e30f;
        Lrow[threadIdx.x] = 0.f;
        for (int c = 0; c < 16; c++) Oacc[threadIdx.x * 16 + c] = 0.f;
    }
    // Prologue DMAs: Q tile + first K/V block (wave 0 only; TENSORcnt per-wave)
    if (w == 0) {
        tdm_load_2d_bf16(Qs, Qb + (size_t)q0 * C, 64, 64, (unsigned)C);
        tdm_load_2d_bf16(Kblk[0], Kb, 64, 64, (unsigned)C);
        tdm_load_2d_bf16(Vraw[0], Vb, 16, 64, (unsigned)Cs);
    }

    int ti = w >> 1, tj0 = (w & 1) * 2;           // 8 waves x 2 tiles = 16 tiles
    int nblk = N / 64;
    for (int ib = 0; ib < nblk; ib++) {
        int buf = ib & 1;
        if (w == 0) __builtin_amdgcn_s_wait_tensorcnt(0);
        __syncthreads();   // buf ready for all; buf^1 reads from ib-1 finished
        // Kick off next block's DMA into the other buffer (overlaps compute)
        if (w == 0 && (ib + 1) < nblk) {
            size_t kb2 = (size_t)(ib + 1) * 64;
            tdm_load_2d_bf16(Kblk[buf ^ 1], Kb + kb2 * C, 64, 64, (unsigned)C);
            tdm_load_2d_bf16(Vraw[buf ^ 1], Vb + kb2 * Cs, 16, 64, (unsigned)Cs);
        }
        // Transpose V block in LDS: Vt[j][k] = Vraw[k][j]
        for (int idx = threadIdx.x; idx < 64 * 16; idx += 256) {
            int kk = idx >> 4, j = idx & 15;
            Vt[j * 64 + kk] = Vraw[buf][kk * 16 + j];
        }
        // S = Q_tile @ K_blk^T from LDS
        f32x8 s0 = {0.f,0.f,0.f,0.f,0.f,0.f,0.f,0.f};
        f32x8 s1 = s0;
        for (int k0 = 0; k0 < 64; k0 += 32) {
            bf16x16 a  = load_a_rm((const bf16_t*)Qs + ti * 16 * 64 + k0, 64);
            bf16x16 b0 = load_b_nk((const bf16_t*)Kblk[buf] + tj0 * 16 * 64 + k0, 64);
            bf16x16 b1 = load_b_nk((const bf16_t*)Kblk[buf] + (tj0 + 1) * 16 * 64 + k0, 64);
            s0 = wmma_bf16(a, b0, s0);
            s1 = wmma_bf16(a, b1, s1);
        }
#pragma unroll
        for (int r = 0; r < 8; r++) {
            int i = ti * 16 + r + hp * 8;
            Sblk[i * 64 + tj0 * 16 + m]       = s0[r];
            Sblk[i * 64 + (tj0 + 1) * 16 + m] = s1[r];
        }
        __syncthreads();
        // Online softmax update per query row
        if (threadIdx.x < 64) {
            int r = threadIdx.x;
            float mo = Mrow[r], mx = mo;
            for (int c = 0; c < 64; c++) {
                float v = Sblk[r * 64 + c] * st;
                Sblk[r * 64 + c] = v;
                mx = fmaxf(mx, v);
            }
            float corr = __expf(mo - mx);
            float lsum = 0.f;
            for (int c = 0; c < 64; c++) {
                float p = __expf(Sblk[r * 64 + c] - mx);
                Pblk[r * 64 + c] = f32_bf16(p);
                lsum += p;
            }
            Lrow[r] = Lrow[r] * corr + lsum;
            Mrow[r] = mx;
            for (int c = 0; c < 16; c++) Oacc[r * 16 + c] *= corr;
        }
        __syncthreads();
        // O += P @ V  (waves 0..3, one 16-row tile each, K=64)
        if (w < 4) {
            f32x8 acc = {0.f,0.f,0.f,0.f,0.f,0.f,0.f,0.f};
            for (int k0 = 0; k0 < 64; k0 += 32) {
                bf16x16 a  = load_a_rm((const bf16_t*)Pblk + w * 16 * 64 + k0, 64);
                bf16x16 bb = load_b_nk((const bf16_t*)Vt + k0, 64);
                acc = wmma_bf16(a, bb, acc);
            }
#pragma unroll
            for (int r = 0; r < 8; r++)
                Oacc[(w * 16 + r + hp * 8) * 16 + m] += acc[r];
        }
        __syncthreads();
    }
    // Epilogue: divide by l, scatter via torch raw reshape into cols 256..319
    if (threadIdx.x < 64) {
        int r = threadIdx.x;
        float inv = 1.f / Lrow[r];
        for (int c = 0; c < 16; c++) {
            float v = Oacc[r * 16 + c] * inv;
            long flat = ((long)hd * N + (q0 + r)) * 16 + c;
            int rp = (int)(flat >> 6);
            int c2 = (int)(flat & 63);
            out[((size_t)b * N + rp) * 320 + 256 + c2] = v;
        }
    }
}

// ---------------------------------------------------------------------------
// Host-side launcher
// ---------------------------------------------------------------------------
extern "C" void kernel_launch(void* const* d_in, const int* in_sizes, int n_in,
                              void* d_out, int out_size, void* d_ws, size_t ws_size,
                              hipStream_t stream) {
    (void)in_sizes; (void)n_in; (void)out_size; (void)ws_size;
    const int B = 4, N = 2048, C = 256, Cs = 64;
    const int BNC = B * N * C;     // 2,097,152
    const int BNCs = B * N * Cs;   // 524,288

    const float* s   = (const float*)d_in[0];
    const float* h   = (const float*)d_in[1];
    const float* sh  = (const float*)d_in[2];
    const float* t1  = (const float*)d_in[3];
    const float* t2  = (const float*)d_in[4];
    const float* Wqc = (const float*)d_in[5];
    const float* Wqs = (const float*)d_in[6];
    const float* Wkc = (const float*)d_in[7];
    const float* Wvc = (const float*)d_in[8];
    const float* Wks = (const float*)d_in[9];
    const float* Wvs = (const float*)d_in[10];
    float* out = (float*)d_out;

    char* ws = (char*)d_ws;
    size_t off = 0;
    auto take = [&](size_t bytes) -> unsigned short* {
        unsigned short* p = (unsigned short*)(ws + off);
        off = (off + bytes + 255) & ~(size_t)255;
        return p;
    };
    unsigned short* s_bf  = take((size_t)BNC * 2);
    unsigned short* sh_bf = take((size_t)BNC * 2);
    unsigned short* h_bf  = take((size_t)BNCs * 2);
    unsigned short* wqc = take((size_t)C * C * 2);
    unsigned short* wqs = take((size_t)C * C * 2);
    unsigned short* wkc = take((size_t)C * C * 2);
    unsigned short* wvc = take((size_t)C * C * 2);
    unsigned short* wks = take((size_t)C * C * 2);
    unsigned short* wvs = take((size_t)Cs * Cs * 2);
    unsigned short* Pqc = take((size_t)BNC * 2);
    unsigned short* Pqs = take((size_t)BNC * 2);
    unsigned short* Pkc = take((size_t)BNC * 2);
    unsigned short* Pvc = take((size_t)BNC * 2);
    unsigned short* Pks = take((size_t)BNC * 2);
    unsigned short* Pvs = take((size_t)BNCs * 2);
    unsigned short* Sca = take((size_t)16 * 64 * 64 * 2);
    // total ~31 MB of workspace

    // 1) f32 -> bf16
    k_cvt_bf16<<<(BNC + 255) / 256, 256, 0, stream>>>(s, s_bf, BNC);
    k_cvt_bf16<<<(BNC + 255) / 256, 256, 0, stream>>>(sh, sh_bf, BNC);
    k_cvt_bf16<<<(BNCs + 255) / 256, 256, 0, stream>>>(h, h_bf, BNCs);
    k_cvt_bf16<<<(C * C + 255) / 256, 256, 0, stream>>>(Wqc, wqc, C * C);
    k_cvt_bf16<<<(C * C + 255) / 256, 256, 0, stream>>>(Wqs, wqs, C * C);
    k_cvt_bf16<<<(C * C + 255) / 256, 256, 0, stream>>>(Wkc, wkc, C * C);
    k_cvt_bf16<<<(C * C + 255) / 256, 256, 0, stream>>>(Wvc, wvc, C * C);
    k_cvt_bf16<<<(C * C + 255) / 256, 256, 0, stream>>>(Wks, wks, C * C);
    k_cvt_bf16<<<(Cs * Cs + 255) / 256, 256, 0, stream>>>(Wvs, wvs, Cs * Cs);

    // 2) projections: x @ W^T  (M = B*N = 8192)
    k_proj_gemm<<<dim3(64, 4), 256, 0, stream>>>(s_bf,  wqc, Pqc, 8192, C, C);
    k_proj_gemm<<<dim3(64, 4), 256, 0, stream>>>(sh_bf, wqs, Pqs, 8192, C, C);
    k_proj_gemm<<<dim3(64, 4), 256, 0, stream>>>(sh_bf, wkc, Pkc, 8192, C, C);
    k_proj_gemm<<<dim3(64, 4), 256, 0, stream>>>(sh_bf, wvc, Pvc, 8192, C, C);
    k_proj_gemm<<<dim3(64, 4), 256, 0, stream>>>(sh_bf, wks, Pks, 8192, C, C);
    k_proj_gemm<<<dim3(64, 1), 256, 0, stream>>>(h_bf,  wvs, Pvs, 8192, Cs, Cs);

    // 3) channel branch
    k_chan_scores<<<16, 256, 0, stream>>>(Pqc, Pkc, t1, Sca, N, C);
    k_chan_av<<<dim3(N / 128, 16), 256, 0, stream>>>(Sca, Pvc, out, N, C);

    // 4) spatial branch (flash attention, TDM-staged)
    k_spatial_flash<<<dim3(N / 64, 16), 256, 0, stream>>>(Pqs, Pks, Pvs, t2,
                                                          out, N, C, Cs);
}
